// AttentionRNNLayer_87677462380995
// MI455X (gfx1250) — matrix operand order repaired and checked by reading, not compiled, not run on hardware
//
#include <hip/hip_runtime.h>
#include <hip/hip_bf16.h>
#include <stdint.h>

// ---------------------------------------------------------------------------
// Problem constants (from reference): B=64, T=512, F=128, U=512
// ---------------------------------------------------------------------------
#define B_   64
#define T_   512
#define F_   128
#define U_   512
#define KTOT 640     // 512 (recurrent h) + 128 (input x), fused K dimension
#define NTOT 2048    // 4 gates * U

typedef __attribute__((ext_vector_type(16))) __bf16 v16bf;
typedef __attribute__((ext_vector_type(8)))  float  v8f;

union Frag16 {                 // 16 bf16 = 32 bytes = two b128 loads
    v16bf          v;
    uint4          q[2];
    unsigned short us[16];
};

__device__ __forceinline__ unsigned short f2bf_rne(float f) {
    unsigned u = __float_as_uint(f);
    unsigned r = u + 0x7FFFu + ((u >> 16) & 1u);   // round-to-nearest-even
    return (unsigned short)(r >> 16);
}

__device__ __forceinline__ float sigf(float x) {
    return 1.0f / (1.0f + __expf(-x));
}

// B-matrix fragment (32x16, 16-bit): lane l holds column N=l%16;
// k order used here (same as A): element e <-> K = k0 + e + 16*(l>=16). Contiguous 16 bf16 per lane.
__device__ __forceinline__ v16bf load_bfrag(const unsigned short* __restrict__ row, int k0) {
    Frag16 b;
    b.q[0] = *(const uint4*)(row + k0);
    b.q[1] = *(const uint4*)(row + k0 + 8);
    return b.v;
}

// ---------------------------------------------------------------------------
// Prep: pack W_comb[n][k] (bf16, transposed, k<512 from Wh, k>=512 from Wx),
// zero h0 / c / barrier counter.  Idempotent per call (graph-replay safe).
// ---------------------------------------------------------------------------
__global__ void lstm_prep(const float* __restrict__ Wx, const float* __restrict__ Wh,
                          unsigned short* __restrict__ Wcomb,
                          unsigned short* __restrict__ h0,
                          float* __restrict__ c0, unsigned* __restrict__ cnt) {
    int idx = blockIdx.x * blockDim.x + threadIdx.x;
    if (idx < NTOT * KTOT) {
        int n = idx / KTOT, k = idx - n * KTOT;
        float v = (k < U_) ? Wh[(size_t)k * NTOT + n] : Wx[(size_t)(k - U_) * NTOT + n];
        Wcomb[idx] = f2bf_rne(v);
    }
    if (idx < B_ * U_) { h0[idx] = 0; c0[idx] = 0.0f; }
    if (idx == 0) *cnt = 0u;
}

// ---------------------------------------------------------------------------
// Persistent LSTM scan: 32 blocks x 128 threads = 128 waves.
// Wave (mt, ut) computes the four 16x16 gate tiles for batch rows
// [16*mt,16*mt+16) and units [16*ut,16*ut+16); K=640 fused reduction
// (h_{t-1} part from bf16 ping-pong buffer, x_t part converted on the fly).
// Grid-wide monotonic-counter barrier between timesteps.
// ---------------------------------------------------------------------------
__global__ void __launch_bounds__(128, 1)
lstm_scan(const float* __restrict__ x, const float* __restrict__ b_lstm,
          const unsigned short* __restrict__ Wcomb,
          unsigned short* __restrict__ hb0, unsigned short* __restrict__ hb1,
          float* __restrict__ cbuf, unsigned* __restrict__ cnt) {
    const int lane = threadIdx.x & 31;
    const int wave = blockIdx.x * (blockDim.x >> 5) + (threadIdx.x >> 5);  // 0..127
    const int mt   = wave & 3;          // 4 m-tiles (batch)
    const int ut   = wave >> 2;         // 32 unit-tiles
    const int l15  = lane & 15;
    const int hi   = (lane >> 4) & 1;
    const int m    = mt * 16 + l15;     // batch row this lane feeds into A
    const int nloc = ut * 16 + l15;     // unit column this lane owns in C/D

    // A-matrix (16x32, 16-bit): lane l holds row M=l%16;
    // k order used here (same as B, not AMD's table order): element e <-> K = k0 + e + 16*(l>=16)
    const int akoff = hi * 16;
    const int bkoff = hi * 16;

    // Per-gate bias: all 8 C elements of a lane share N=nloc -> 1 scalar/gate,
    // invariant over time: hoist out of the scan.
    const float bi = b_lstm[0 * U_ + nloc];
    const float bf = b_lstm[1 * U_ + nloc];
    const float bg = b_lstm[2 * U_ + nloc];
    const float bo = b_lstm[3 * U_ + nloc];

    const float* xrow = x + (size_t)m * (T_ * F_);
    const unsigned short* wr0 = Wcomb + (size_t)(0 * U_ + nloc) * KTOT;
    const unsigned short* wr1 = Wcomb + (size_t)(1 * U_ + nloc) * KTOT;
    const unsigned short* wr2 = Wcomb + (size_t)(2 * U_ + nloc) * KTOT;
    const unsigned short* wr3 = Wcomb + (size_t)(3 * U_ + nloc) * KTOT;

    for (int t = 0; t < T_; ++t) {
        const unsigned short* __restrict__ hprev = (t & 1) ? hb1 : hb0;
        unsigned short* __restrict__ hnext       = (t & 1) ? hb0 : hb1;

        v8f acc0 = {}, acc1 = {}, acc2 = {}, acc3 = {};

        // ---- recurrent reduction: K in [0,512) over h_{t-1} (bf16 buffer)
        const unsigned short* hrow = hprev + (size_t)m * U_;
        for (int kc = 0; kc < U_; kc += 32) {
            Frag16 a;
            a.q[0] = *(const uint4*)(hrow + kc + akoff);        // e=0..7
            a.q[1] = *(const uint4*)(hrow + kc + akoff + 8);   // e=8..15
            v16bf b0 = load_bfrag(wr0, kc + bkoff);
            acc0 = __builtin_amdgcn_wmma_f32_16x16x32_bf16(false, a.v, false, b0, (short)0, acc0, false, false);
            v16bf b1 = load_bfrag(wr1, kc + bkoff);
            acc1 = __builtin_amdgcn_wmma_f32_16x16x32_bf16(false, a.v, false, b1, (short)0, acc1, false, false);
            v16bf b2 = load_bfrag(wr2, kc + bkoff);
            acc2 = __builtin_amdgcn_wmma_f32_16x16x32_bf16(false, a.v, false, b2, (short)0, acc2, false, false);
            v16bf b3 = load_bfrag(wr3, kc + bkoff);
            acc3 = __builtin_amdgcn_wmma_f32_16x16x32_bf16(false, a.v, false, b3, (short)0, acc3, false, false);
        }

        // ---- fused input reduction: K in [512,640) over x[:, t, :] (f32 -> bf16)
        const float* xt = xrow + (size_t)t * F_;
        for (int kx = 0; kx < F_; kx += 32) {
            float4 f0 = *(const float4*)(xt + kx + akoff);
            float4 f1 = *(const float4*)(xt + kx + akoff + 4);
            float4 f2 = *(const float4*)(xt + kx + akoff + 8);
            float4 f3 = *(const float4*)(xt + kx + akoff + 12);
            Frag16 a;
            a.us[0]  = f2bf_rne(f0.x); a.us[1]  = f2bf_rne(f0.y);
            a.us[2]  = f2bf_rne(f0.z); a.us[3]  = f2bf_rne(f0.w);
            a.us[4]  = f2bf_rne(f1.x); a.us[5]  = f2bf_rne(f1.y);
            a.us[6]  = f2bf_rne(f1.z); a.us[7]  = f2bf_rne(f1.w);
            a.us[8]  = f2bf_rne(f2.x); a.us[9]  = f2bf_rne(f2.y);
            a.us[10] = f2bf_rne(f2.z); a.us[11] = f2bf_rne(f2.w);
            a.us[12] = f2bf_rne(f3.x); a.us[13] = f2bf_rne(f3.y);
            a.us[14] = f2bf_rne(f3.z); a.us[15] = f2bf_rne(f3.w);
            const int kb = U_ + kx + bkoff;
            v16bf b0 = load_bfrag(wr0, kb);
            acc0 = __builtin_amdgcn_wmma_f32_16x16x32_bf16(false, a.v, false, b0, (short)0, acc0, false, false);
            v16bf b1 = load_bfrag(wr1, kb);
            acc1 = __builtin_amdgcn_wmma_f32_16x16x32_bf16(false, a.v, false, b1, (short)0, acc1, false, false);
            v16bf b2 = load_bfrag(wr2, kb);
            acc2 = __builtin_amdgcn_wmma_f32_16x16x32_bf16(false, a.v, false, b2, (short)0, acc2, false, false);
            v16bf b3 = load_bfrag(wr3, kb);
            acc3 = __builtin_amdgcn_wmma_f32_16x16x32_bf16(false, a.v, false, b3, (short)0, acc3, false, false);
        }

        // ---- gate nonlinearity in C-layout registers (M = j + 8*hi, N = nloc)
        #pragma unroll
        for (int j = 0; j < 8; ++j) {
            const int mrow = mt * 16 + j + 8 * hi;
            const size_t off = (size_t)mrow * U_ + nloc;
            float zi = acc0[j] + bi;
            float zf = acc1[j] + bf;
            float zg = acc2[j] + bg;
            float zo = acc3[j] + bo;
            float cn = sigf(zf) * cbuf[off] + sigf(zi) * tanhf(zg);
            cbuf[off]  = cn;
            hnext[off] = f2bf_rne(sigf(zo) * tanhf(cn));
        }

        // ---- grid-wide barrier (monotonic counter; all 32 blocks co-resident)
        __threadfence();
        __syncthreads();
        if (threadIdx.x == 0) {
            atomicAdd(cnt, 1u);
            const unsigned target = (unsigned)gridDim.x * (unsigned)(t + 1);
            while (atomicAdd(cnt, 0u) < target) __builtin_amdgcn_s_sleep(2);
        }
        __syncthreads();
        __threadfence();
    }
}

// ---------------------------------------------------------------------------
// Reference tail: d=(B,T,1); softmax over the singleton axis == 1.0;
// out = sigmoid(1.0) — a constant, independent of everything upstream.
// ---------------------------------------------------------------------------
__global__ void fill_out(float* __restrict__ out, int n) {
    int i = blockIdx.x * blockDim.x + threadIdx.x;
    if (i < n) out[i] = 0.73105857863000487925f;  // sigmoid(1.0)
}

extern "C" void kernel_launch(void* const* d_in, const int* in_sizes, int n_in,
                              void* d_out, int out_size, void* d_ws, size_t ws_size,
                              hipStream_t stream) {
    const float* x      = (const float*)d_in[0];  // (B,T,F)
    const float* Wx     = (const float*)d_in[1];  // (F,4U)
    const float* Wh     = (const float*)d_in[2];  // (U,4U)
    const float* b_lstm = (const float*)d_in[3];  // (4U,)
    // d_in[4..8] (Wa, ba, ua, Wd, bd) are provably dead: softmax over a
    // size-1 axis makes the output a constant.

    // Workspace layout (256B-aligned slices)
    const size_t szW  = (size_t)NTOT * KTOT * sizeof(unsigned short); // 2,621,440
    const size_t szH  = (size_t)B_ * U_ * sizeof(unsigned short);     //    65,536
    const size_t szC  = (size_t)B_ * U_ * sizeof(float);              //   131,072
    size_t off = 0;
    char* ws = (char*)d_ws;
    unsigned short* Wcomb = (unsigned short*)(ws + off); off += szW;
    unsigned short* hb0   = (unsigned short*)(ws + off); off += szH;
    unsigned short* hb1   = (unsigned short*)(ws + off); off += szH;
    float*          cbuf  = (float*)(ws + off);          off += szC;
    unsigned*       cnt   = (unsigned*)(ws + off);       off += 256;

    if (ws_size >= off) {
        const int prepN = NTOT * KTOT;                 // 1,310,720 elements
        lstm_prep<<<(prepN + 255) / 256, 256, 0, stream>>>(Wx, Wh, Wcomb, hb0, cbuf, cnt);
        // (hb0 zeroed above; hb1 is write-before-read at t=0)
        lstm_scan<<<32, 128, 0, stream>>>(x, b_lstm, Wcomb, hb0, hb1, cbuf, cnt);
    }

    fill_out<<<(out_size + 255) / 256, 256, 0, stream>>>((float*)d_out, out_size);
    (void)in_sizes; (void)n_in;
}
